// Conv2d_62182536512134
// MI455X (gfx1250) — compile-verified, matrix operand order checked by reading, not run on hardware
//
#include <hip/hip_runtime.h>
#include <stdint.h>

// ---------------------------------------------------------------------------
// StyleGAN2 fused upsample2x -> conv3x3 -> FIR(1,3,3,1) -> +bias  (NCHW, fp32)
//
// Kernel 1: implicit-GEMM transposed conv via v_wmma_f32_16x16x32_f16.
//   Data staged into LDS by the Tensor Data Mover (tensor_load_to_lds,
//   TENSORcnt-tracked), fragments built from LDS (ds_load + cvt).
//   Per WG: one padded row p, 64 y-columns, 32 out-channels.
//   K = input channels (8 chunks of 32), accumulated over the 9 conv taps
//   with per-lane parity/bounds predication (zero-insertion upsampling).
// Kernel 2: depthwise 4x4 FIR (separable 1,3,3,1 / 16) + per-channel bias.
// ---------------------------------------------------------------------------

typedef __attribute__((ext_vector_type(16))) _Float16     v16h;
typedef __attribute__((ext_vector_type(8)))  float        v8f;
typedef __attribute__((ext_vector_type(4)))  unsigned int u32x4;
typedef __attribute__((ext_vector_type(4)))  int          i32x4;
typedef __attribute__((ext_vector_type(8)))  int          i32x8;

#define NB   16
#define IC   256
#define OC   256
#define IH   64
#define IW   64
#define YP   131          // padded intermediate height/width (129 + 1 + 1)
#define OH   128
#define OW   128

struct SMem {
    float xrow[2][32][IW];    // [kh-slot][ic_local][iw]        16 KB
    float wt[32][288];        // [oc_local][ic_local*9 + tap]   36 KB
};

// Tensor Data Mover: 2-D tile load global->LDS.
//   tile: tile0 contiguous elements per line, tile1 lines, line stride
//   stride0 (in 4-byte elements). D# packed per CDNA5 ISA.
__device__ __forceinline__ void tdm_load_2d(unsigned lds_off, const void* gptr,
                                            unsigned tile0, unsigned tile1,
                                            unsigned long stride0,
                                            unsigned tdim0, unsigned tdim1)
{
    unsigned long ga = (unsigned long)(uintptr_t)gptr;

    u32x4 g0;
    g0[0] = 1u;                                           // count=1 (user D#)
    g0[1] = lds_off;                                      // lds_addr (bytes)
    g0[2] = (unsigned)(ga & 0xFFFFFFFFu);                 // global_addr[31:0]
    g0[3] = (unsigned)((ga >> 32) & 0x1FFFFFFu)           // global_addr[56:32]
          | (2u << 30);                                   // type=2 ("image")

    i32x8 g1;
    g1[0] = (int)(2u << 16);                              // data_size=4B
    g1[1] = (int)((tdim0 & 0xFFFFu) << 16);               // tensor_dim0[15:0]
    g1[2] = (int)(((tdim0 >> 16) & 0xFFFFu)
          | ((tdim1 & 0xFFFFu) << 16));                   // dim0[31:16]|dim1[15:0]
    g1[3] = (int)(((tdim1 >> 16) & 0xFFFFu)
          | ((tile0 & 0xFFFFu) << 16));                   // dim1[31:16]|tile_dim0
    g1[4] = (int)(tile1 & 0xFFFFu);                       // tile_dim1 (tile_dim2=0)
    g1[5] = (int)(stride0 & 0xFFFFFFFFu);                 // tensor_dim0_stride lo
    g1[6] = (int)((stride0 >> 32) & 0xFFFFu);             // stride hi (dim1_stride=0)
    g1[7] = 0;

    i32x4 z4 = {0, 0, 0, 0};
#if __clang_major__ >= 23
    i32x8 z8 = {0, 0, 0, 0, 0, 0, 0, 0};
    __builtin_amdgcn_tensor_load_to_lds(g0, g1, z4, z4, z8, 0);
#else
    __builtin_amdgcn_tensor_load_to_lds(g0, g1, z4, z4, 0);
#endif
}

__global__ __launch_bounds__(256)
void upconv_wmma_kernel(const float* __restrict__ x,
                        const float* __restrict__ w,
                        float* __restrict__ yp)
{
    __shared__ SMem sm;

    // grid: x = q-tile (3 x 64 cols), y = row*8 + octile (131*8), z = batch
    const int qtile  = blockIdx.x;          // 0..2
    const int row_oc = blockIdx.y;
    const int p      = row_oc >> 3;         // padded row, 0..130
    const int octile = row_oc & 7;          // 0..7 (32 ocs each)
    const int b      = blockIdx.z;

    const int tid     = threadIdx.x;
    const int wave    = tid >> 5;           // 8 waves
    const int lane    = tid & 31;
    const int qchunk  = wave & 3;           // 4 position chunks of 16
    const int occhunk = wave >> 2;          // 2 oc chunks of 16

    const int half = lane >> 4;
    const int m    = lane & 15;             // A: row M ; B: column N

    const int q  = qtile * 64 + qchunk * 16 + m;
    const int pp = p - 1;
    const int qq = q - 1;
    const bool interior = (pp >= 0) & (pp < 129) & (qq >= 0) & (qq < 129);

    const int oc0 = octile * 32;
    const int ocl = occhunk * 16 + m;       // lane's B column within 32-oc tile

    // the two input rows any output row p can touch:
    //   kh==0   -> row (pp-2)/2   (slot 0)
    //   kh==1,2 -> row  pp/2      (slot 1)
    int ih0 = (pp - 2) >> 1; ih0 = ih0 < 0 ? 0 : (ih0 > IH - 1 ? IH - 1 : ih0);
    int ih1 = pp >> 1;       ih1 = ih1 < 0 ? 0 : (ih1 > IH - 1 ? IH - 1 : ih1);

    v8f c = {0.f, 0.f, 0.f, 0.f, 0.f, 0.f, 0.f, 0.f};

    for (int kc = 0; kc < 8; ++kc) {
        const int ic0 = kc * 32;

        __syncthreads();                    // LDS free to overwrite
        if (wave == 0) {
            // x rows: 64 contiguous floats per ic, 32 ics, stride 4096 elems
            tdm_load_2d((unsigned)(uintptr_t)&sm.xrow[0][0][0],
                        x + ((long)(b * IC + ic0) * (IH * IW) + ih0 * IW),
                        IW, 32, (unsigned long)(IH * IW), IW, 32);
            tdm_load_2d((unsigned)(uintptr_t)&sm.xrow[1][0][0],
                        x + ((long)(b * IC + ic0) * (IH * IW) + ih1 * IW),
                        IW, 32, (unsigned long)(IH * IW), IW, 32);
            // weights: 288 contiguous floats (32 ic x 9 tap) per oc, 32 ocs,
            // stride 2304 elems between ocs
            tdm_load_2d((unsigned)(uintptr_t)&sm.wt[0][0],
                        w + ((long)(oc0 * IC + ic0) * 9),
                        288, 32, (unsigned long)(IC * 9), 288, 32);
            __builtin_amdgcn_s_wait_tensorcnt(0);
        }
        __syncthreads();                    // tiles visible to all waves

        for (int tap = 0; tap < 9; ++tap) {
            const int kh = tap / 3;
            const int kw = tap % 3;
            const int ph = pp + kh - 2;
            const int qw = qq + kw - 2;
            const int iw = qw >> 1;
            const bool ok = interior & (ph >= 0) & (qw >= 0) &
                            ((ph & 1) == 0) & ((qw & 1) == 0) &
                            ((ph >> 1) < IH) & (iw < IW);
            const int slot = (kh == 0) ? 0 : 1;

            v16h a, bm;
            // 16-bit A 16x32 layout: lanes<16 -> K in {0..7,16..23},
            //                        lanes>=16 -> K in {8..15,24..31}
            #pragma unroll
            for (int j = 0; j < 8; ++j) {
                const int kbase = ((j >= 4) ? 16 : 0) + half * 8 + (j & 3) * 2;
                #pragma unroll
                for (int e = 0; e < 2; ++e) {
                    const int k = kbase + e;
                    const float xv = ok ? sm.xrow[slot][k][iw] : 0.0f;
                    a[2 * j + e]  = (_Float16)xv;
                    bm[2 * j + e] = (_Float16)sm.wt[ocl][k * 9 + tap];
                }
            }

            c = __builtin_amdgcn_wmma_f32_16x16x32_f16(
                    false, a, false, bm, (short)0, c, false, false);
        }
    }

    // C/D layout: lane column N = lane&15 (oc), VGPR r holds M = r + 8*half.
    const int ocs = oc0 + occhunk * 16 + (lane & 15);
    #pragma unroll
    for (int r = 0; r < 8; ++r) {
        const int ms = r + 8 * half;
        const int qs = qtile * 64 + qchunk * 16 + ms;
        if (qs < YP) {
            const long yidx = (((long)(b * OC + ocs) * YP) + p) * YP + qs;
            yp[yidx] = c[r];
        }
    }
}

__global__ __launch_bounds__(256)
void fir_bias_kernel(const float* __restrict__ yp,
                     const float* __restrict__ bias,
                     float* __restrict__ out)
{
    const long idx   = (long)blockIdx.x * blockDim.x + threadIdx.x;
    const long total = (long)NB * OC * OH * OW;
    if (idx >= total) return;

    const int  nn = (int)(idx & (OW - 1));
    const int  mm = (int)((idx >> 7) & (OH - 1));
    const long bc = idx >> 14;                 // b*256 + c
    const int  ch = (int)(bc & (OC - 1));

    const float f1[4] = {1.f, 3.f, 3.f, 1.f};  // separable FIR (symmetric)
    const float* base = yp + bc * (YP * YP) + (long)mm * YP + nn;

    float acc = 0.f;
    #pragma unroll
    for (int u = 0; u < 4; ++u) {
        float rs = 0.f;
        #pragma unroll
        for (int v = 0; v < 4; ++v)
            rs += f1[v] * base[u * YP + v];
        acc += f1[u] * rs;
    }
    out[idx] = acc * (1.0f / 16.0f) + bias[ch];
}

extern "C" void kernel_launch(void* const* d_in, const int* in_sizes, int n_in,
                              void* d_out, int out_size, void* d_ws, size_t ws_size,
                              hipStream_t stream)
{
    const float* x  = (const float*)d_in[0];   // (16,256,64,64) f32
    const float* w  = (const float*)d_in[1];   // (256,256,3,3)  f32
    const float* bs = (const float*)d_in[2];   // (256,)         f32
    float* out = (float*)d_out;                // (16,256,128,128) f32
    float* yp  = (float*)d_ws;                 // (16,256,131,131) f32 padded y

    dim3 gA(3, YP * 8, NB);
    upconv_wmma_kernel<<<gA, 256, 0, stream>>>(x, w, yp);

    const long total = (long)NB * OC * OH * OW;
    const int blocks = (int)((total + 255) / 256);
    fir_bias_kernel<<<blocks, 256, 0, stream>>>(yp, bs, out);
}
